// Token_Selective_Attention_52888227283095
// MI455X (gfx1250) — matrix operand order repaired and checked by reading, not compiled, not run on hardware
//
#include <hip/hip_runtime.h>
#include <hip/hip_bf16.h>
#include <stdint.h>

#define HEADS 8
#define GROUPS 4
#define HD 24
#define WD 24
#define DIM 1024
#define CS 256              // DIM/GROUPS
#define CHN 32              // CS/HEADS
#define NTOK 2304           // HD*WD*GROUPS
#define HW 576
#define TOPK 1843           // int(2304*0.8)
#define NT (NTOK/16)        // 144 n/m tiles
#define MGROUPS 18          // 144/8
#define KSTEPS_PV (NTOK/32) // 72
#define KSTEPS_PR (DIM/32)  // 32

typedef __attribute__((ext_vector_type(16))) __bf16   v16bf;
typedef __attribute__((ext_vector_type(8)))  float    v8f;
typedef __attribute__((ext_vector_type(4)))  uint32_t v4u;

union Frag { v16bf v; v4u u[2]; };

__device__ __forceinline__ __bf16 f2bf(float f) {
  union { float f; uint32_t u; } in; in.f = f;
  uint32_t u = in.u;
  uint32_t r = (u + 0x7fffu + ((u >> 16) & 1u)) >> 16;   // round-to-nearest-even
  uint16_t h = (uint16_t)r;
  __bf16 o; __builtin_memcpy(&o, &h, 2);
  return o;
}

// ---------------------------------------------------------------------------
// K1: fused group-mix (12x4) + depthwise 3x3 conv, scattering straight into
// head-major layouts: qf/kf f32 [head][ch][n], vb bf16 [head][ch][n]
// ---------------------------------------------------------------------------
__global__ void k_qkv_dwconv(const float* __restrict__ x,
                             const float* __restrict__ wqkv,
                             const float* __restrict__ wdw,
                             float* __restrict__ qf, float* __restrict__ kf,
                             __bf16* __restrict__ vb) {
  int idx = blockIdx.x * blockDim.x + threadIdx.x;   // 12*256*576 threads
  int s   = idx / (CS * HW);
  int rem = idx % (CS * HW);
  int c   = rem / HW;
  int pix = rem % HW;
  int hh = pix / WD, ww = pix % WD;
  float acc = 0.f;
  #pragma unroll
  for (int dy = -1; dy <= 1; ++dy) {
    int yy = hh + dy;
    if ((unsigned)yy >= HD) continue;
    #pragma unroll
    for (int dx = -1; dx <= 1; ++dx) {
      int xx = ww + dx;
      if ((unsigned)xx >= WD) continue;
      float pre = 0.f;
      #pragma unroll
      for (int g = 0; g < GROUPS; ++g)
        pre += wqkv[s * GROUPS + g] * x[((g * CS + c) * HD + yy) * WD + xx];
      acc += wdw[s * 9 + (dy + 1) * 3 + (dx + 1)] * pre;
    }
  }
  int g = s & 3, kind = s >> 2;                // s: 0..3 q, 4..7 k, 8..11 v
  int head = c >> 5, ch = c & 31;
  int n = pix * GROUPS + g;                    // n = (hh*W+ww)*t + g
  int off = (head * CHN + ch) * NTOK + n;
  if      (kind == 0) qf[off] = acc;
  else if (kind == 1) kf[off] = acc;
  else                vb[off] = f2bf(acc);
}

// ---------------------------------------------------------------------------
// K2: L2-normalize q/k rows over N=2304 and transpose to WMMA-A-friendly
// bf16 layout [head][n][32] (64B contiguous rows)
// ---------------------------------------------------------------------------
__global__ void k_normalize(const float* __restrict__ qf, const float* __restrict__ kf,
                            __bf16* __restrict__ qt, __bf16* __restrict__ kt) {
  __shared__ float red[256];
  int which = blockIdx.x >> 8;           // 0=q 1=k
  int row   = blockIdx.x & 255;          // head*32+ch
  int head = row >> 5, ch = row & 31;
  const float* src = (which ? kf : qf) + row * NTOK;
  __bf16* dst = which ? kt : qt;
  float ss = 0.f;
  for (int i = threadIdx.x; i < NTOK; i += 256) { float v = src[i]; ss += v * v; }
  red[threadIdx.x] = ss; __syncthreads();
  for (int o = 128; o > 0; o >>= 1) {
    if (threadIdx.x < o) red[threadIdx.x] += red[threadIdx.x + o];
    __syncthreads();
  }
  float scale = 1.f / fmaxf(sqrtf(red[0]), 1e-12f);
  for (int i = threadIdx.x; i < NTOK; i += 256)
    dst[(head * NTOK + i) * CHN + ch] = f2bf(src[i] * scale);
}

// ---------------------------------------------------------------------------
// K3: attn = (q^T k) * temperature. K=32 exactly -> one wmma per 16x16 tile.
// A fragment reused across 8 m-tiles per wave.
// ---------------------------------------------------------------------------
__global__ void k_qk_wmma(const __bf16* __restrict__ qt, const __bf16* __restrict__ kt,
                          const float* __restrict__ temp, float* __restrict__ attn) {
  int wid = threadIdx.x >> 5, lane = threadIdx.x & 31;
  int wg   = blockIdx.x * 8 + wid;            // 8*144*18 waves
  int head = wg / (NT * MGROUPS);
  int rem  = wg % (NT * MGROUPS);
  int ntile = rem / MGROUPS;
  int mg    = rem % MGROUPS;
  int half = lane >> 4, l15 = lane & 15;
  const __bf16* qh = qt + head * NTOK * CHN;
  const __bf16* kh = kt + head * NTOK * CHN;
  Frag a;
  {
    const __bf16* rp = qh + (ntile * 16 + l15) * CHN + half * 8;
    a.u[0] = *(const v4u*)(rp);          // K = kb+0..7
    a.u[1] = *(const v4u*)(rp + 16);     // K = kb+16..23
  }
  float tscale = temp[head];
  float* ah = attn + (size_t)head * NTOK * NTOK;
  #pragma unroll
  for (int m = 0; m < 8; ++m) {
    int mtile = mg * 8 + m;
    Frag b;
    const __bf16* rp = kh + (mtile * 16 + l15) * CHN + half * 8;
    b.u[0] = *(const v4u*)(rp);          // K = kb+0..7
    b.u[1] = *(const v4u*)(rp + 16);     // K = kb+16..23
    v8f c = {};
    c = __builtin_amdgcn_wmma_f32_16x16x32_bf16(false, a.v, false, b.v,
                                                (short)0, c, false, false);
    #pragma unroll
    for (int r = 0; r < 8; ++r) {
      int rr = ntile * 16 + r + half * 8;
      int cc = mtile * 16 + l15;
      ah[(size_t)rr * NTOK + cc] = c[r] * tscale;
    }
  }
}

// ---------------------------------------------------------------------------
// K4: per-row top-k threshold (bisection on count) + masked softmax -> bf16 p
// One 256-thread block per row. The row is staged memory->LDS with gfx1250
// async copies (GLOBAL_LOAD_ASYNC_TO_LDS_B128, tracked by ASYNCcnt): no VGPR
// round trip, wave can sleep on s_wait_asynccnt while the 9KB row lands.
// ---------------------------------------------------------------------------
__global__ void k_topk_softmax(const float* __restrict__ attn, __bf16* __restrict__ p) {
  __shared__ float srow[NTOK];
  __shared__ float redf[256];
  __shared__ int   redi[256];
  size_t rbase = (size_t)blockIdx.x * NTOK;  // 8*2304 rows
  const float* src = attn + rbase;
  int tid = threadIdx.x;

  // --- async stage: 576 x 16B chunks, direct to LDS ---
  {
    uint32_t lds0 = (uint32_t)(uintptr_t)(&srow[0]);
    for (int chunk = tid; chunk < NTOK / 4; chunk += 256) {
      uint32_t laddr = lds0 + (uint32_t)chunk * 16u;
      uint64_t gaddr = (uint64_t)(uintptr_t)(src + chunk * 4);
      asm volatile("global_load_async_to_lds_b128 %0, %1, off"
                   :: "v"(laddr), "v"(gaddr) : "memory");
    }
    asm volatile("s_wait_asynccnt 0x0" ::: "memory");
  }
  __syncthreads();

  float lmax = -__builtin_inff(), lmin = __builtin_inff();
  for (int i = tid; i < NTOK; i += 256) {
    float v = srow[i];
    lmax = fmaxf(lmax, v); lmin = fminf(lmin, v);
  }
  redf[tid] = lmax; __syncthreads();
  for (int o = 128; o > 0; o >>= 1) { if (tid < o) redf[tid] = fmaxf(redf[tid], redf[tid + o]); __syncthreads(); }
  float rmax = redf[0]; __syncthreads();
  redf[tid] = lmin; __syncthreads();
  for (int o = 128; o > 0; o >>= 1) { if (tid < o) redf[tid] = fminf(redf[tid], redf[tid + o]); __syncthreads(); }
  float rmin = redf[0]; __syncthreads();
  float lo = rmin, hi = rmax;                // invariant: count(>=lo) >= K
  for (int it = 0; it < 30; ++it) {
    float mid = 0.5f * (lo + hi);
    int cnt = 0;
    for (int i = tid; i < NTOK; i += 256) cnt += (srow[i] >= mid) ? 1 : 0;
    redi[tid] = cnt; __syncthreads();
    for (int o = 128; o > 0; o >>= 1) { if (tid < o) redi[tid] += redi[tid + o]; __syncthreads(); }
    int total = redi[0]; __syncthreads();
    if (total >= TOPK) lo = mid; else hi = mid;
  }
  float thr = lo;
  float lsum = 0.f;
  for (int i = tid; i < NTOK; i += 256) {
    float v = srow[i];
    if (v >= thr) lsum += __expf(v - rmax);
  }
  redf[tid] = lsum; __syncthreads();
  for (int o = 128; o > 0; o >>= 1) { if (tid < o) redf[tid] += redf[tid + o]; __syncthreads(); }
  float inv = 1.f / redf[0];
  __bf16* dst = p + rbase;
  for (int i = tid; i < NTOK; i += 256) {
    float v = srow[i];
    dst[i] = (v >= thr) ? f2bf(__expf(v - rmax) * inv) : f2bf(0.f);
  }
}

// ---------------------------------------------------------------------------
// K5: out = p @ v^T  (M=2304 tokens, N=32 channels, K=2304), accumulate in C.
// Prefetch the next K-step of the dominant p stream (global_prefetch_b8).
// Result scattered into channel-major bf16 layout outhw[hw][c_full] for proj.
// ---------------------------------------------------------------------------
__global__ void k_pv_wmma(const __bf16* __restrict__ p, const __bf16* __restrict__ vb,
                          __bf16* __restrict__ outhw) {
  int wid = threadIdx.x >> 5, lane = threadIdx.x & 31;
  int wg   = blockIdx.x * 8 + wid;          // 8*144*2 waves
  int head = wg / (NT * 2);
  int rem  = wg % (NT * 2);
  int ntile = rem / 2;
  int ctile = rem % 2;
  int half = lane >> 4, l15 = lane & 15;
  const __bf16* prow = p  + ((size_t)head * NTOK + ntile * 16 + l15) * NTOK;
  const __bf16* vrow = vb + ((size_t)head * CHN  + ctile * 16 + l15) * NTOK;
  v8f acc = {};
  for (int kk = 0; kk < KSTEPS_PV; ++kk) {
    int mk = kk * 32;
    Frag a, b;
    const __bf16* ap = prow + mk + half * 8;
    a.u[0] = *(const v4u*)(ap);
    a.u[1] = *(const v4u*)(ap + 16);
    const __bf16* bp = vrow + mk + half * 8;
    b.u[0] = *(const v4u*)(bp);
    b.u[1] = *(const v4u*)(bp + 16);
    if (kk + 1 < KSTEPS_PV)
      __builtin_prefetch(prow + mk + 32, 0, 0);   // next p K-step
    acc = __builtin_amdgcn_wmma_f32_16x16x32_bf16(false, a.v, false, b.v,
                                                  (short)0, acc, false, false);
  }
  #pragma unroll
  for (int r = 0; r < 8; ++r) {
    int n = ntile * 16 + r + half * 8;
    int c = ctile * 16 + l15;
    int hw = n >> 2, g = n & 3;
    int cf = g * CS + head * CHN + c;       // full channel index
    outhw[hw * DIM + cf] = f2bf(acc[r]);
  }
}

__global__ void k_cvt_bf16(const float* __restrict__ src, __bf16* __restrict__ dst, int n) {
  int i = blockIdx.x * blockDim.x + threadIdx.x;
  if (i < n) dst[i] = f2bf(src[i]);
}

// ---------------------------------------------------------------------------
// K6: y = x + W_proj @ out  (M=1024, N=576, K=1024) + residual, f32 out.
// ---------------------------------------------------------------------------
__global__ void k_proj_wmma(const __bf16* __restrict__ wpb, const __bf16* __restrict__ outhw,
                            const float* __restrict__ x, float* __restrict__ y) {
  int wid = threadIdx.x >> 5, lane = threadIdx.x & 31;
  int wg = blockIdx.x * 8 + wid;            // 64*36 waves
  int otile = wg / 36;
  int htile = wg % 36;
  int half = lane >> 4, l15 = lane & 15;
  const __bf16* arow = wpb   + (otile * 16 + l15) * DIM;
  const __bf16* brow = outhw + (htile * 16 + l15) * DIM;
  v8f acc = {};
  for (int kk = 0; kk < KSTEPS_PR; ++kk) {
    int ck = kk * 32;
    Frag a, b;
    const __bf16* ap = arow + ck + half * 8;
    a.u[0] = *(const v4u*)(ap);
    a.u[1] = *(const v4u*)(ap + 16);
    const __bf16* bp = brow + ck + half * 8;
    b.u[0] = *(const v4u*)(bp);
    b.u[1] = *(const v4u*)(bp + 16);
    acc = __builtin_amdgcn_wmma_f32_16x16x32_bf16(false, a.v, false, b.v,
                                                  (short)0, acc, false, false);
  }
  #pragma unroll
  for (int r = 0; r < 8; ++r) {
    int o  = otile * 16 + r + half * 8;
    int hw = htile * 16 + l15;
    y[o * HW + hw] = x[o * HW + hw] + acc[r];
  }
}

extern "C" void kernel_launch(void* const* d_in, const int* in_sizes, int n_in,
                              void* d_out, int out_size, void* d_ws, size_t ws_size,
                              hipStream_t stream) {
  (void)in_sizes; (void)n_in; (void)out_size; (void)ws_size;
  const float* x     = (const float*)d_in[0];
  const float* temp  = (const float*)d_in[1];
  const float* wqkv  = (const float*)d_in[2];
  const float* wdw   = (const float*)d_in[3];
  const float* wproj = (const float*)d_in[4];
  float* out = (float*)d_out;

  char* ws = (char*)d_ws;
  size_t off = 0;
  auto alloc = [&](size_t bytes) -> void* {
    off = (off + 255) & ~(size_t)255;
    void* pp = ws + off;
    off += bytes;
    return pp;
  };
  float*  qf    = (float*) alloc((size_t)HEADS * CHN * NTOK * 4);
  float*  kf    = (float*) alloc((size_t)HEADS * CHN * NTOK * 4);
  __bf16* vb    = (__bf16*)alloc((size_t)HEADS * CHN * NTOK * 2);
  __bf16* qt    = (__bf16*)alloc((size_t)HEADS * NTOK * CHN * 2);
  __bf16* kt    = (__bf16*)alloc((size_t)HEADS * NTOK * CHN * 2);
  __bf16* wpb   = (__bf16*)alloc((size_t)DIM * DIM * 2);
  __bf16* outhw = (__bf16*)alloc((size_t)HW * DIM * 2);
  __bf16* pmat  = (__bf16*)alloc((size_t)HEADS * NTOK * NTOK * 2);
  float*  attn  = (float*) alloc((size_t)HEADS * NTOK * NTOK * 4);

  k_qkv_dwconv<<<(3 * GROUPS * CS * HW) / 256, 256, 0, stream>>>(x, wqkv, wdw, qf, kf, vb);
  k_cvt_bf16<<<(DIM * DIM) / 256, 256, 0, stream>>>(wproj, wpb, DIM * DIM);
  k_normalize<<<2 * HEADS * CHN, 256, 0, stream>>>(qf, kf, qt, kt);
  k_qk_wmma<<<(HEADS * NT * MGROUPS) / 8, 256, 0, stream>>>(qt, kt, temp, attn);
  k_topk_softmax<<<HEADS * NTOK, 256, 0, stream>>>(attn, pmat);
  k_pv_wmma<<<(HEADS * NT * 2) / 8, 256, 0, stream>>>(pmat, vb, outhw);
  k_proj_wmma<<<(64 * 36) / 8, 256, 0, stream>>>(wpb, outhw, x, out);
}
